// ResidualVectorQuantizationWithClustering_489626272395
// MI455X (gfx1250) — matrix operand order repaired and checked by reading, not compiled, not run on hardware
//
#include <hip/hip_runtime.h>

// CDNA5 (gfx1250) wave32 WMMA types
typedef __attribute__((ext_vector_type(16))) __bf16 v16bf;
typedef __attribute__((ext_vector_type(8)))  float  v8f;

#define NROWS 262144
#define DIMS  64
#define LEVELS 4
#define KCENT 1024
#define ROWS_PER_BLOCK 128   // 8 waves * 16 rows
#define TILES_PER_CHUNK 16   // 16 column tiles staged in LDS at a time (32 KB)
#define NCHUNKS (KCENT / 16 / TILES_PER_CHUNK) // 4

// A-matrix (16-bit, 16x32) per-lane element -> K mapping (ISA 7.12.2):
// lane half h = lane>>4, element e in v16bf, operand o (K base 0 or 32):
//   K = o*32 + h*8 + (e & 7) + (e >= 8 ? 16 : 0)
__device__ __forceinline__ int kmapA(int o, int h, int e) {
    return o * 32 + h * 8 + (e & 7) + ((e >> 3) << 4);
}

__global__ __launch_bounds__(256)
void rvq_wmma_kernel(const float* __restrict__ feat,
                     const float* __restrict__ cb,
                     float* __restrict__ outq,
                     int*   __restrict__ outIdx)
{
    // LDS: staged bf16 B operands (WMMA-B layout), center norms, per-wave idx scratch
    __shared__ __align__(32) __bf16 s_B[TILES_PER_CHUNK * 2 * 32 * 16]; // 32 KB
    __shared__ float s_cnorm[KCENT];                                    // 4 KB
    __shared__ int   s_idx[8 * 16];                                     // 512 B

    const int tid  = threadIdx.x;
    const int lane = tid & 31;
    const int wave = tid >> 5;
    const int h    = lane >> 4;   // lane half
    const int m    = lane & 15;   // row (A) or column (B/C) within tile
    const int rowGlobal = blockIdx.x * ROWS_PER_BLOCK + wave * 16 + m;

    // Load this lane's slice of its feature row directly in WMMA-A f32 mirror
    // layout. Pairs (even e, e+1) are contiguous K -> 8-byte loads.
    float resf[32];   // residual, A layout:  j = o*16 + e
    float finit[32];  // initial feature (for telescoped output)
    {
        const float* frow = feat + (size_t)rowGlobal * DIMS;
#pragma unroll
        for (int j = 0; j < 32; j += 2) {
            const int e = j & 15, o = j >> 4;
            const int k = kmapA(o, h, e);
            const float2 v = *(const float2*)(frow + k);
            resf[j] = v.x; resf[j + 1] = v.y;
            finit[j] = v.x; finit[j + 1] = v.y;
        }
    }

    for (int l = 0; l < LEVELS; ++l) {
        const float* cbl = cb + (size_t)l * KCENT * DIMS;

        __syncthreads(); // everyone done reading previous level's LDS
        // center norms ||c_k||^2 (argmin needs cnorm - 2*dot; row norm drops out)
        for (int c = tid; c < KCENT; c += 256) {
            const float* cr = cbl + c * DIMS;
            float s = 0.f;
#pragma unroll
            for (int d = 0; d < DIMS; ++d) s = fmaf(cr[d], cr[d], s);
            s_cnorm[c] = s;
        }

        // Convert residual to bf16 WMMA-A operands (K=0..31 and K=32..63)
        v16bf a0, a1;
#pragma unroll
        for (int e = 0; e < 16; ++e) {
            a0[e] = (__bf16)resf[e];
            a1[e] = (__bf16)resf[16 + e];
        }

        // running per-row-vgpr argmin state: f32 score + i32 column.
        // Strictly-less update + increasing column order per lane => first-min
        // tie-break (argmin semantics) within a lane; cross-lane ties handled
        // in the butterfly via an ordered u64 key.
        float    minS[8];
        unsigned minI[8];
#pragma unroll
        for (int v = 0; v < 8; ++v) { minS[v] = __builtin_inff(); minI[v] = 0u; }

        for (int c = 0; c < NCHUNKS; ++c) {
            if (c > 0) __syncthreads(); // waves done scanning previous chunk
            // Stage 16 tiles * 2 operands of B into LDS as bf16, WMMA-B layout.
            // B layout: lane n=lane&15, half hh=lane>>4, element e -> K = o*32 + hh*8 + (e & 7) + (e >= 8 ? 16 : 0)
            // => each (tile,o,lane) row is two 8-float runs of center n.
#pragma unroll
            for (int r = 0; r < 4; ++r) {
                const int rowId = tid * 4 + r;          // 0..1023
                const int bl = rowId & 31;
                const int o  = (rowId >> 5) & 1;
                const int tt = rowId >> 6;              // tile within chunk
                const int n  = (c * TILES_PER_CHUNK + tt) * 16 + (bl & 15);
                const int hh = bl >> 4;
                const float* src = cbl + n * DIMS + o * 32 + hh * 8;
                v16bf tv;
#pragma unroll
                for (int e = 0; e < 16; ++e) tv[e] = (__bf16)src[(e & 7) + ((e >> 3) << 4)];
                *(v16bf*)(s_B + (size_t)rowId * 16) = tv;
            }
            __syncthreads(); // staged B (and, for c==0, cnorm) visible

            // Software-pipelined tile scan: issue WMMAs for tile t, then do the
            // argmin update for tile t-1 while tile t's WMMA is in flight. We
            // never read the most-recent WMMA dest, so the 4-coexec-NOP
            // WMMA->VALU RAW hazard slots are filled with useful VALU work.
            v8f accP;                 // pipelined accumulator (tile t-1)
            float cnP;
            unsigned colP;
            {   // prologue: tile 0 of this chunk
                const int t = c * TILES_PER_CHUNK;
                const v16bf b0 = *(const v16bf*)(s_B + (0 * 32 + lane) * 16);
                const v16bf b1 = *(const v16bf*)(s_B + (1 * 32 + lane) * 16);
                cnP = s_cnorm[t * 16 + m];
                v8f acc = {0.f, 0.f, 0.f, 0.f, 0.f, 0.f, 0.f, 0.f};
                acc = __builtin_amdgcn_wmma_f32_16x16x32_bf16(
                        false, a0, false, b0, (short)0, acc, false, false);
                accP = __builtin_amdgcn_wmma_f32_16x16x32_bf16(
                        false, a1, false, b1, (short)0, acc, false, false);
                colP = (unsigned)(t * 16 + m);
            }
#pragma unroll 5
            for (int tt = 1; tt < TILES_PER_CHUNK; ++tt) {
                const int t = c * TILES_PER_CHUNK + tt;
                const v16bf b0 = *(const v16bf*)(s_B + ((tt * 2 + 0) * 32 + lane) * 16);
                const v16bf b1 = *(const v16bf*)(s_B + ((tt * 2 + 1) * 32 + lane) * 16);
                const float cn = s_cnorm[t * 16 + m];
                v8f acc = {0.f, 0.f, 0.f, 0.f, 0.f, 0.f, 0.f, 0.f};
                acc = __builtin_amdgcn_wmma_f32_16x16x32_bf16(
                        false, a0, false, b0, (short)0, acc, false, false);
                acc = __builtin_amdgcn_wmma_f32_16x16x32_bf16(
                        false, a1, false, b1, (short)0, acc, false, false);
                // consume previous tile's accumulator (long since complete)
#pragma unroll
                for (int v = 0; v < 8; ++v) {
                    const float score = fmaf(-2.0f, accP[v], cnP);
                    const bool lt = score < minS[v];   // v_cmp + dual cndmask
                    minS[v] = lt ? score : minS[v];
                    minI[v] = lt ? colP  : minI[v];
                }
                accP = acc;
                cnP  = cn;
                colP = (unsigned)(t * 16 + m);
            }
            {   // epilogue: flush last tile of this chunk
#pragma unroll
                for (int v = 0; v < 8; ++v) {
                    const float score = fmaf(-2.0f, accP[v], cnP);
                    const bool lt = score < minS[v];
                    minS[v] = lt ? score : minS[v];
                    minI[v] = lt ? colP  : minI[v];
                }
            }
        }

        // Build ordered u64 keys once (monotone float->uint map; low word = col)
        // then butterfly-min across the 16-lane column group. Group 0
        // (lanes 0-15) holds rows 0..7, group 1 (lanes 16-31) rows 8..15.
        unsigned long long keys[8];
#pragma unroll
        for (int v = 0; v < 8; ++v) {
            unsigned u = __float_as_uint(minS[v]);
            u = (u & 0x80000000u) ? ~u : (u | 0x80000000u);
            keys[v] = ((unsigned long long)u << 32) | (unsigned long long)minI[v];
        }
#pragma unroll
        for (int off = 8; off >= 1; off >>= 1) {
#pragma unroll
            for (int v = 0; v < 8; ++v) {
                unsigned lo = (unsigned)keys[v];
                unsigned hi = (unsigned)(keys[v] >> 32);
                lo = (unsigned)__shfl_xor((int)lo, off, 32);
                hi = (unsigned)__shfl_xor((int)hi, off, 32);
                const unsigned long long other =
                    ((unsigned long long)hi << 32) | (unsigned long long)lo;
                if (other < keys[v]) keys[v] = other;
            }
        }

        // lanes 0..7 publish rows 0..7; lanes 16..23 publish rows 8..15
        if (m < 8) {
            const int rowInWave = h * 8 + m;
            const int idx = (int)(keys[m] & 0xFFFFFFFFull);
            const int row = blockIdx.x * ROWS_PER_BLOCK + wave * 16 + rowInWave;
            outIdx[(size_t)l * NROWS + row] = idx;
            s_idx[wave * 16 + rowInWave] = idx;
        }
        // wave-local LDS RAW across lanes: drain DS counter explicitly (CDNA5)
        asm volatile("s_wait_dscnt 0" ::: "memory");
        const int myIdx = s_idx[wave * 16 + m];

        // exact f32 gather + residual update: r -= centers[idx]
        const float* crow = cbl + (size_t)myIdx * DIMS;
#pragma unroll
        for (int j = 0; j < 32; j += 2) {
            const int e = j & 15, o = j >> 4;
            const int k = kmapA(o, h, e);
            const float2 v = *(const float2*)(crow + k);
            resf[j]     -= v.x;
            resf[j + 1] -= v.y;
        }
    }

    // sum of quantized levels telescopes: sum(q_l) = f_init - r_final
    float* orow = outq + (size_t)rowGlobal * DIMS;
#pragma unroll
    for (int j = 0; j < 32; j += 2) {
        const int e = j & 15, o = j >> 4;
        const int k = kmapA(o, h, e);
        float2 v;
        v.x = finit[j]     - resf[j];
        v.y = finit[j + 1] - resf[j + 1];
        *(float2*)(orow + k) = v;
    }
}

extern "C" void kernel_launch(void* const* d_in, const int* in_sizes, int n_in,
                              void* d_out, int out_size, void* d_ws, size_t ws_size,
                              hipStream_t stream) {
    const float* feat = (const float*)d_in[0];  // [N, 64] f32
    const float* cb   = (const float*)d_in[1];  // [4, 1024, 64] f32
    float* outq   = (float*)d_out;                                // [N, 64] f32
    int*   outIdx = (int*)((float*)d_out + (size_t)NROWS * DIMS); // [4, N] i32

    rvq_wmma_kernel<<<NROWS / ROWS_PER_BLOCK, 256, 0, stream>>>(feat, cb, outq, outIdx);
}
